// DecoderBlock_85237920956537
// MI455X (gfx1250) — matrix operand order repaired and checked by reading, not compiled, not run on hardware
//
#include <hip/hip_runtime.h>
#include <math.h>

// ---------------- problem constants (from reference) ----------------
#define DD    1024          // D
#define DI2   2048          // DI
#define NSZ   16            // NS
#define DTRK  64            // DTR
#define KCV   4             // KC
#define NE    8             // E
#define DFFN  2048          // DFF
#define BB    2
#define TT    2048
#define NTOK  (BB*TT)       // 4096 tokens
#define PROJW 96            // DTR + 2*NS (true width)
#define PROJS 128           // padded proj stride / padded x_proj rows

typedef __attribute__((ext_vector_type(16))) __bf16 v16bf;
typedef __attribute__((ext_vector_type(8)))  float  v8f;
typedef __attribute__((ext_vector_type(4)))  int    v4i;

// async global->LDS DMA (gfx1250), guarded so missing builtin falls back cleanly
#if defined(__has_builtin)
#  if __has_builtin(__builtin_amdgcn_global_load_async_to_lds_b128)
#    define HAVE_ASYNC_LDS 1
#  endif
#endif
#ifndef HAVE_ASYNC_LDS
#  define HAVE_ASYNC_LDS 0
#endif

#if HAVE_ASYNC_LDS
typedef __attribute__((address_space(1))) v4i as1_v4i;  // global int4
typedef __attribute__((address_space(3))) v4i as3_v4i;  // LDS int4
#if __has_builtin(__builtin_amdgcn_s_wait_asynccnt)
#  define WAIT_ASYNC() __builtin_amdgcn_s_wait_asynccnt(0)
#else
#  define WAIT_ASYNC() asm volatile("s_wait_asynccnt 0x0" ::: "memory")
#endif
#define NBUF 2
#else
#define NBUF 1
#endif

__device__ __forceinline__ unsigned short f2bf(float f) {
  __bf16 h = (__bf16)f;
  return __builtin_bit_cast(unsigned short, h);
}
__device__ __forceinline__ float siluf(float x) { return x / (1.0f + expf(-x)); }

#define WMMA_BF16(acc_, a_, b_)                                                     \
  acc_ = __builtin_amdgcn_wmma_f32_16x16x32_bf16(false, a_, false, b_, (short)0,    \
                                                 acc_, false, false)

// A fragment 16x32: lane m=l16; element j -> K = base + half*16 + j
__device__ __forceinline__ v16bf load_afrag(const unsigned short* p, int half) {
  v16bf a;
#pragma unroll
  for (int j = 0; j < 16; ++j) {
    int kk = half * 16 + j;
    a[j] = __builtin_bit_cast(__bf16, p[kk]);
  }
  return a;
}
// B fragment 32x16: lane n=l16; 16 contiguous K starting at half*16 (p pre-offset)
__device__ __forceinline__ v16bf load_bfrag(const unsigned short* p) {
  v16bf b;
#pragma unroll
  for (int j = 0; j < 16; ++j) b[j] = __builtin_bit_cast(__bf16, p[j]);
  return b;
}

// ---------------- fp32 -> bf16 casts ----------------
__global__ void castf2bf(const float* __restrict__ s, unsigned short* __restrict__ d, size_t n) {
  size_t i = (size_t)blockIdx.x * blockDim.x + threadIdx.x;
  size_t st = (size_t)gridDim.x * blockDim.x;
  for (; i < n; i += st) d[i] = f2bf(s[i]);
}

__global__ void copyf(const float* __restrict__ s, float* __restrict__ d, size_t n) {
  size_t i = (size_t)blockIdx.x * blockDim.x + threadIdx.x;
  size_t st = (size_t)gridDim.x * blockDim.x;
  for (; i < n; i += st) d[i] = s[i];
}

// x_proj weight (96 x DI2) -> bf16 zero-padded to (128 x DI2)
__global__ void cast_pad_xproj(const float* __restrict__ s, unsigned short* __restrict__ d) {
  int i = blockIdx.x * 256 + threadIdx.x;
  if (i >= PROJS * DI2) return;
  int r = i / DI2, c = i % DI2;
  d[i] = (r < PROJW) ? f2bf(s[(size_t)r * DI2 + c]) : (unsigned short)0;
}

// dt_r = proj[:, 0:64] -> bf16 (proj stride PROJS)
__global__ void cast_dtr(const float* __restrict__ proj, unsigned short* __restrict__ d) {
  int i = blockIdx.x * 256 + threadIdx.x;
  if (i < NTOK * DTRK) {
    int m = i >> 6, j = i & 63;
    d[i] = f2bf(proj[(size_t)m * PROJS + j]);
  }
}

// ---------------- LayerNorm (one block per token, D=1024) ----------------
__global__ __launch_bounds__(256)
void ln_kernel(const float* __restrict__ x, const float* __restrict__ g,
               const float* __restrict__ bta, unsigned short* __restrict__ obf,
               float* __restrict__ of) {
  __shared__ float s1[256], s2[256];
  const int tok = blockIdx.x;
  const float* row = x + (size_t)tok * DD;
  float v[4], sum = 0.f, sq = 0.f;
#pragma unroll
  for (int i = 0; i < 4; ++i) { v[i] = row[threadIdx.x + i * 256]; sum += v[i]; sq += v[i] * v[i]; }
  s1[threadIdx.x] = sum; s2[threadIdx.x] = sq;
  __syncthreads();
  for (int st = 128; st > 0; st >>= 1) {
    if (threadIdx.x < st) { s1[threadIdx.x] += s1[threadIdx.x + st]; s2[threadIdx.x] += s2[threadIdx.x + st]; }
    __syncthreads();
  }
  float mu  = s1[0] * (1.0f / DD);
  float var = s2[0] * (1.0f / DD) - mu * mu;
  float rs  = rsqrtf(var + 1e-5f);
#pragma unroll
  for (int i = 0; i < 4; ++i) {
    int c = threadIdx.x + i * 256;
    float o = (v[i] - mu) * rs * g[c] + bta[c];
    obf[(size_t)tok * DD + c] = f2bf(o);
    if (of) of[(size_t)tok * DD + c] = o;
  }
}

// ---------------- bf16 WMMA GEMM: C = A(MxK) * W(NoutxK)^T ----------------
// 256 threads = 8 waves in 4(M) x 2(N); block tile 128x64; wave tile 32x32 (2x2 wmma).
// BK=64 (two 16x16x32 k-substeps); double-buffered async global->LDS when available.
// modes: 0=store acc(+bias); 1=silu->bf16; 2=residual Cf=acc+R;
//        3=Cf += gate[m*NE+gi]*(acc+bias); 4=softplus
#define BM 128
#define BN 64
#define BK 64
#define LDSA 72   // row stride (bf16) for A tile: 16B aligned, conflict-free
#define LDSB 72
__global__ __launch_bounds__(256)
void gemm_bf16_wmma(const unsigned short* __restrict__ A, const unsigned short* __restrict__ Bw,
                    const float* __restrict__ bias, float* __restrict__ Cf,
                    unsigned short* __restrict__ Cbf, const float* __restrict__ R,
                    const float* __restrict__ gate, int M, int Nout, int K,
                    int mode, int gateIdx) {
  __shared__ unsigned short lA[NBUF][BM * LDSA];
  __shared__ unsigned short lB[NBUF][BN * LDSB];
  const int tid  = threadIdx.x;
  const int lane = tid & 31, wave = tid >> 5;
  const int wm = wave & 3, wn = wave >> 2;     // 4(M) x 2(N) waves
  const int half = lane >> 4, l16 = lane & 15;
  const int n0 = blockIdx.x * BN;
  const int m0 = blockIdx.y * BM;

  v8f acc[2][2] = {};

#if HAVE_ASYNC_LDS
  auto stage_async = [&](int k0, int buf) {
#pragma unroll
    for (int i = 0; i < 4; ++i) {                 // A: 128 rows x 8 chunks(8 bf16)
      int task = tid + i * 256;
      int r = task >> 3, c = (task & 7) * 8;
      __builtin_amdgcn_global_load_async_to_lds_b128(
          (as1_v4i*)(v4i*)(A + (size_t)(m0 + r) * K + k0 + c),
          (as3_v4i*)(v4i*)&lA[buf][r * LDSA + c], 0, 0);
    }
#pragma unroll
    for (int i = 0; i < 2; ++i) {                 // B: 64 rows x 8 chunks
      int task = tid + i * 256;
      int r = task >> 3, c = (task & 7) * 8;
      __builtin_amdgcn_global_load_async_to_lds_b128(
          (as1_v4i*)(v4i*)(Bw + (size_t)(n0 + r) * K + k0 + c),
          (as3_v4i*)(v4i*)&lB[buf][r * LDSB + c], 0, 0);
    }
  };
#else
  auto stage_sync = [&](int k0, int buf) {
#pragma unroll
    for (int i = 0; i < 4; ++i) {
      int task = tid + i * 256;
      int r = task >> 3, c = (task & 7) * 8;
      *reinterpret_cast<uint4*>(&lA[buf][r * LDSA + c]) =
          *reinterpret_cast<const uint4*>(A + (size_t)(m0 + r) * K + k0 + c);
    }
#pragma unroll
    for (int i = 0; i < 2; ++i) {
      int task = tid + i * 256;
      int r = task >> 3, c = (task & 7) * 8;
      *reinterpret_cast<uint4*>(&lB[buf][r * LDSB + c]) =
          *reinterpret_cast<const uint4*>(Bw + (size_t)(n0 + r) * K + k0 + c);
    }
    if (k0 + BK < K) {
      __builtin_prefetch(A + (size_t)(m0 + (tid >> 3)) * K + k0 + BK, 0, 1);
      __builtin_prefetch(Bw + (size_t)(n0 + (tid >> 3 & 63)) * K + k0 + BK, 0, 1);
    }
  };
#endif

  auto compute = [&](int buf) {
    const unsigned short* pa = lA[buf];
    const unsigned short* pb = lB[buf];
#pragma unroll
    for (int ks = 0; ks < BK; ks += 32) {
      v16bf a0 = load_afrag(&pa[(wm * 32 + 0  + l16) * LDSA + ks], half);
      v16bf a1 = load_afrag(&pa[(wm * 32 + 16 + l16) * LDSA + ks], half);
      v16bf b0 = load_bfrag(&pb[(wn * 32 + 0  + l16) * LDSB + ks + half * 16]);
      v16bf b1 = load_bfrag(&pb[(wn * 32 + 16 + l16) * LDSB + ks + half * 16]);
      WMMA_BF16(acc[0][0], a0, b0);
      WMMA_BF16(acc[0][1], a0, b1);
      WMMA_BF16(acc[1][0], a1, b0);
      WMMA_BF16(acc[1][1], a1, b1);
    }
  };

  const int nkt = K / BK;
#if HAVE_ASYNC_LDS
  stage_async(0, 0);
  WAIT_ASYNC();
  __syncthreads();
  for (int kt = 0; kt < nkt; ++kt) {
    const int cur = kt & 1;
    if (kt + 1 < nkt) stage_async((kt + 1) * BK, cur ^ 1);  // overlap DMA w/ WMMA
    compute(cur);
    WAIT_ASYNC();
    __syncthreads();
  }
#else
  for (int kt = 0; kt < nkt; ++kt) {
    __syncthreads();
    stage_sync(kt * BK, 0);
    __syncthreads();
    compute(0);
  }
#endif

  // epilogue: D layout per 16x16 tile — VGPR r: M = r + half*8, N = l16
#pragma unroll
  for (int ni = 0; ni < 2; ++ni) {
    const int n = n0 + wn * 32 + ni * 16 + l16;
    const float bn = bias ? bias[n] : 0.0f;
#pragma unroll
    for (int mi = 0; mi < 2; ++mi) {
      const v8f av = acc[mi][ni];
#pragma unroll
      for (int r = 0; r < 8; ++r) {
        const int m = m0 + wm * 32 + mi * 16 + half * 8 + r;
        float v = av[r] + bn;
        const size_t off = (size_t)m * Nout + n;
        if (mode == 0) {
          Cf[off] = v;
        } else if (mode == 1) {
          float s = siluf(v);
          if (Cf) Cf[off] = s;
          Cbf[off] = f2bf(s);
        } else if (mode == 2) {
          Cf[off] = v + R[off];
        } else if (mode == 3) {
          Cf[off] += gate[(size_t)m * NE + gateIdx] * v;
        } else { // 4: softplus
          Cf[off] = (v > 20.0f) ? v : log1pf(expf(v));
        }
      }
    }
  }
}

// ---------------- causal depthwise conv (KC=4) + SiLU ----------------
__global__ void conv_silu(const float* __restrict__ xz, const float* __restrict__ cw,
                          const float* __restrict__ cb, float* __restrict__ xc,
                          unsigned short* __restrict__ xcbf) {
  int i = blockIdx.x * 256 + threadIdx.x;
  if (i >= NTOK * DI2) return;
  int tok = i / DI2, d = i % DI2;
  int b = tok / TT, t = tok % TT;
  float acc = cb[d];
#pragma unroll
  for (int k = 0; k < KCV; ++k) {
    int tc = t + k - (KCV - 1);
    if (tc >= 0) acc += cw[d * KCV + k] * xz[(size_t)(b * TT + tc) * (2 * DI2) + d];
  }
  float s = siluf(acc);
  xc[i] = s;
  xcbf[i] = f2bf(s);
}

// ---------------- selective scan: thread per (batch, channel), 16 states in regs ----------------
#define TCH 64
__global__ __launch_bounds__(256)
void ssm_scan(const float* __restrict__ xz, const float* __restrict__ xc,
              const float* __restrict__ dt, const float* __restrict__ proj,
              const float* __restrict__ A_log, const float* __restrict__ Dsk,
              unsigned short* __restrict__ ybf) {
  __shared__ float sBC[TCH * 32];
  const int b = blockIdx.x / (DI2 / 256);
  const int d = (blockIdx.x % (DI2 / 256)) * 256 + threadIdx.x;
  float Ac[NSZ], h[NSZ];
#pragma unroll
  for (int n = 0; n < NSZ; ++n) { Ac[n] = -expf(A_log[d * NSZ + n]); h[n] = 0.f; }
  const float Dv = Dsk[d];
  for (int t0 = 0; t0 < TT; t0 += TCH) {
    __syncthreads();
#pragma unroll
    for (int i = 0; i < (TCH * 32) / 256; ++i) {
      int idx = threadIdx.x + i * 256;
      int tt = idx >> 5, j = idx & 31;
      sBC[idx] = proj[(size_t)(b * TT + t0 + tt) * PROJS + DTRK + j];
    }
    __syncthreads();
    for (int tt = 0; tt < TCH; ++tt) {
      const size_t tok = (size_t)b * TT + t0 + tt;
      const float dtt = dt[tok * DI2 + d];
      const float xt  = xc[tok * DI2 + d];
      const float dx  = dtt * xt;
      const float* bc = &sBC[tt * 32];
      float y = 0.f;
#pragma unroll
      for (int n = 0; n < NSZ; ++n) {
        float dA = expf(dtt * Ac[n]);
        h[n] = dA * h[n] + dx * bc[n];
        y += h[n] * bc[16 + n];
      }
      const float z = xz[tok * (2 * DI2) + DI2 + d];
      const float yf = (y + xt * Dv) * siluf(z);
      ybf[tok * DI2 + d] = f2bf(yf);
    }
  }
}

// ---------------- router: softmax over E=8, top-2 normalized gates ----------------
__global__ void router_topk(const float* __restrict__ hm, const float* __restrict__ rw,
                            float* __restrict__ probs, float* __restrict__ gate) {
  int tok = blockIdx.x * 256 + threadIdx.x;
  if (tok >= NTOK) return;
  const float* hrow = hm + (size_t)tok * DD;
  float l[NE];
  for (int e = 0; e < NE; ++e) {
    const float* w = rw + (size_t)e * DD;
    float s = 0.f;
    for (int i = 0; i < DD; ++i) s += hrow[i] * w[i];
    l[e] = s;
  }
  float mx = l[0];
#pragma unroll
  for (int e = 1; e < NE; ++e) mx = fmaxf(mx, l[e]);
  float den = 0.f;
#pragma unroll
  for (int e = 0; e < NE; ++e) { l[e] = expf(l[e] - mx); den += l[e]; }
  float inv = 1.0f / den;
  int i1 = 0; float v1 = -1.f;
#pragma unroll
  for (int e = 0; e < NE; ++e) {
    l[e] *= inv;
    probs[(size_t)tok * NE + e] = l[e];
    if (l[e] > v1) { v1 = l[e]; i1 = e; }
  }
  int i2 = -1; float v2 = -1.f;
#pragma unroll
  for (int e = 0; e < NE; ++e) if (e != i1 && l[e] > v2) { v2 = l[e]; i2 = e; }
  float gs = 1.0f / (v1 + v2);
#pragma unroll
  for (int e = 0; e < NE; ++e)
    gate[(size_t)tok * NE + e] = (e == i1) ? v1 * gs : ((e == i2) ? v2 * gs : 0.f);
}

// ---------------- host orchestration ----------------
extern "C" void kernel_launch(void* const* d_in, const int* in_sizes, int n_in,
                              void* d_out, int out_size, void* d_ws, size_t ws_size,
                              hipStream_t stream) {
  const float* x          = (const float*)d_in[0];
  const float* lnm_g      = (const float*)d_in[1];
  const float* lnm_b      = (const float*)d_in[2];
  const float* lnmoe_g    = (const float*)d_in[3];
  const float* lnmoe_b    = (const float*)d_in[4];
  const float* in_proj_w  = (const float*)d_in[5];
  const float* in_proj_b  = (const float*)d_in[6];
  const float* conv_w     = (const float*)d_in[7];
  const float* conv_b     = (const float*)d_in[8];
  const float* x_proj_w   = (const float*)d_in[9];
  const float* dt_proj_w  = (const float*)d_in[10];
  const float* dt_proj_b  = (const float*)d_in[11];
  const float* A_log      = (const float*)d_in[12];
  const float* D_skip     = (const float*)d_in[13];
  const float* out_proj_w = (const float*)d_in[14];
  const float* router_w   = (const float*)d_in[15];
  const float* w1         = (const float*)d_in[16];
  const float* b1         = (const float*)d_in[17];
  const float* w2         = (const float*)d_in[18];
  const float* b2         = (const float*)d_in[19];

  float* out_h     = (float*)d_out;
  float* out_probs = out_h + (size_t)NTOK * DD;

  char* base = (char*)d_ws;
  size_t off = 0;
  auto alloc = [&](size_t bytes) -> void* {
    off = (off + 255) & ~(size_t)255;
    void* p = base + off;
    off += bytes;
    return p;
  };
  typedef unsigned short us;

  us*    wInp  = (us*)alloc((size_t)2 * DI2 * DD * 2);
  us*    wXp   = (us*)alloc((size_t)PROJS * DI2 * 2);      // padded 128 rows
  us*    wDt   = (us*)alloc((size_t)DI2 * DTRK * 2);
  us*    wOut  = (us*)alloc((size_t)DD * DI2 * 2);
  us*    wW1   = (us*)alloc((size_t)NE * DFFN * DD * 2);
  us*    wW2   = (us*)alloc((size_t)NE * DD * DFFN * 2);
  us*    hnbf  = (us*)alloc((size_t)NTOK * DD * 2);
  float* xzf   = (float*)alloc((size_t)NTOK * 2 * DI2 * 4);
  float* xcf   = (float*)alloc((size_t)NTOK * DI2 * 4);
  us*    xcbf  = (us*)alloc((size_t)NTOK * DI2 * 2);
  float* projf = (float*)alloc((size_t)NTOK * PROJS * 4);  // padded stride
  us*    dtrbf = (us*)alloc((size_t)NTOK * DTRK * 2);
  float* dtf   = (float*)alloc((size_t)NTOK * DI2 * 4);
  us*    ybf   = (us*)alloc((size_t)NTOK * DI2 * 2);
  float* hf    = (float*)alloc((size_t)NTOK * DD * 4);
  float* hmf   = (float*)alloc((size_t)NTOK * DD * 4);
  us*    hmbf  = (us*)alloc((size_t)NTOK * DD * 2);
  float* gatef = (float*)alloc((size_t)NTOK * NE * 4);
  us*    h1bf  = (us*)alloc((size_t)NTOK * DFFN * 2);

  auto cast = [&](const float* s, us* d, size_t n) {
    int blocks = (int)((n + 1023) / 1024);
    if (blocks > 2048) blocks = 2048;
    castf2bf<<<blocks, 256, 0, stream>>>(s, d, n);
  };
  cast(in_proj_w,  wInp, (size_t)2 * DI2 * DD);
  cast_pad_xproj<<<(PROJS * DI2) / 256, 256, 0, stream>>>(x_proj_w, wXp);
  cast(dt_proj_w,  wDt,  (size_t)DI2 * DTRK);
  cast(out_proj_w, wOut, (size_t)DD * DI2);
  cast(w1,         wW1,  (size_t)NE * DFFN * DD);
  cast(w2,         wW2,  (size_t)NE * DD * DFFN);

  // Mamba branch
  ln_kernel<<<NTOK, 256, 0, stream>>>(x, lnm_g, lnm_b, hnbf, nullptr);
  gemm_bf16_wmma<<<dim3((2 * DI2) / BN, NTOK / BM), 256, 0, stream>>>(
      hnbf, wInp, in_proj_b, xzf, nullptr, nullptr, nullptr, NTOK, 2 * DI2, DD, 0, 0);
  conv_silu<<<(NTOK * DI2) / 256, 256, 0, stream>>>(xzf, conv_w, conv_b, xcf, xcbf);
  gemm_bf16_wmma<<<dim3(PROJS / BN, NTOK / BM), 256, 0, stream>>>(
      xcbf, wXp, nullptr, projf, nullptr, nullptr, nullptr, NTOK, PROJS, DI2, 0, 0);
  cast_dtr<<<(NTOK * DTRK) / 256, 256, 0, stream>>>(projf, dtrbf);
  gemm_bf16_wmma<<<dim3(DI2 / BN, NTOK / BM), 256, 0, stream>>>(
      dtrbf, wDt, dt_proj_b, dtf, nullptr, nullptr, nullptr, NTOK, DI2, DTRK, 4, 0);
  ssm_scan<<<BB * (DI2 / 256), 256, 0, stream>>>(xzf, xcf, dtf, projf, A_log, D_skip, ybf);
  gemm_bf16_wmma<<<dim3(DD / BN, NTOK / BM), 256, 0, stream>>>(
      ybf, wOut, nullptr, hf, nullptr, x, nullptr, NTOK, DD, DI2, 2, 0);

  // MoE branch
  ln_kernel<<<NTOK, 256, 0, stream>>>(hf, lnmoe_g, lnmoe_b, hmbf, hmf);
  router_topk<<<(NTOK + 255) / 256, 256, 0, stream>>>(hmf, router_w, out_probs, gatef);
  copyf<<<2048, 256, 0, stream>>>(hf, out_h, (size_t)NTOK * DD);
  for (int e = 0; e < NE; ++e) {
    gemm_bf16_wmma<<<dim3(DFFN / BN, NTOK / BM), 256, 0, stream>>>(
        hmbf, wW1 + (size_t)e * DFFN * DD, b1 + (size_t)e * DFFN,
        nullptr, h1bf, nullptr, nullptr, NTOK, DFFN, DD, 1, 0);
    gemm_bf16_wmma<<<dim3(DD / BN, NTOK / BM), 256, 0, stream>>>(
        h1bf, wW2 + (size_t)e * DD * DFFN, b2 + (size_t)e * DD,
        out_h, nullptr, nullptr, gatef, NTOK, DD, DFFN, 3, e);
  }
}
